// StandardAttention_56513179680993
// MI455X (gfx1250) — matrix operand order repaired and checked by reading, not compiled, not run on hardware
//
#include <hip/hip_runtime.h>
#include <hip/hip_bf16.h>
#include <math.h>

typedef __attribute__((ext_vector_type(16))) _Float16 v16h;
typedef __attribute__((ext_vector_type(8)))  _Float16 v8h;
typedef __attribute__((ext_vector_type(8)))  float    v8f;

#define D_MODEL 2048
#define NH      32
#define NKV     8
#define HD      64
#define SEQ     2048
#define BATCH   2
#define NTOK    (BATCH*SEQ)      // 4096
#define QKVN    3072             // 2048 (Q) + 512 (K) + 512 (V)

#define WMMA_F16(A,B,C) __builtin_amdgcn_wmma_f32_16x16x32_f16(false,(A),false,(B),(short)0,(C),false,false)

// ---- fragment loaders, same contiguous k order (k = 16*half + s) in A and B ----
// A 16x32 (MxK): lanes 0-15 row m0+lane, K 0..15; lanes 16-31 same rows, K 16..31
static __device__ __forceinline__ v16h load_a16(const _Float16* base, int ld, int m0, int k0, int lane) {
  int row  = m0 + (lane & 15);
  int hoff = (lane >> 4) << 4;
  const _Float16* p = base + (size_t)row * ld + k0 + hoff;
  return *(const v16h*)p;
}

// B 32x16 (KxN), loaded from row-major B^T [N][K]: lane = column; lanes 0-15 K 0..15, lanes 16-31 K 16..31
static __device__ __forceinline__ v16h load_b16(const _Float16* baseT, int ld, int n0, int k0, int lane) {
  int col  = n0 + (lane & 15);
  int koff = (lane >> 4) << 4;
  return *(const v16h*)(baseT + (size_t)col * ld + k0 + koff);
}

// ---------------- prep kernels ----------------
__global__ void f32_to_f16_kernel(const float* __restrict__ src, _Float16* __restrict__ dst, long n) {
  long i = (long)blockIdx.x * blockDim.x + threadIdx.x;
  if (i < n) dst[i] = (_Float16)src[i];
}

// src [K][N] f32 -> dst [N][K] f16 (LDS-tiled)
__global__ void transpose_f32_f16(const float* __restrict__ src, _Float16* __restrict__ dst, int K, int N) {
  __shared__ float tile[32][33];
  int k0 = blockIdx.y * 32, n0 = blockIdx.x * 32;
  int tx = threadIdx.x;
#pragma unroll
  for (int i = threadIdx.y; i < 32; i += 8)
    tile[i][tx] = src[(size_t)(k0 + i) * N + n0 + tx];
  __syncthreads();
#pragma unroll
  for (int i = threadIdx.y; i < 32; i += 8)
    dst[(size_t)(n0 + i) * K + k0 + tx] = (_Float16)tile[tx][i];
}

// RoPE + layout: qkv [NTOK][3072] -> Qr [B,NH,S,64] (rope), Kr [B,NKV,S,64] (rope), Vt [B,NKV,64,S]
__global__ void rope_scatter(const _Float16* __restrict__ qkv,
                             _Float16* __restrict__ Qr,
                             _Float16* __restrict__ Kr,
                             _Float16* __restrict__ Vt) {
  long idx = (long)blockIdx.x * blockDim.x + threadIdx.x;
  if (idx >= (long)NTOK * QKVN) return;
  int c   = (int)(idx % QKVN);
  int tok = (int)(idx / QKVN);
  int b = tok / SEQ, s = tok % SEQ;
  float v = (float)qkv[idx];
  if (c < NH * HD) {                       // Q path, rope
    int h = c / HD, j = c % HD;
    int jp = (j < 32) ? j + 32 : j - 32;
    float vp = (float)qkv[(size_t)tok * QKVN + h * HD + jp];
    int i = (j < 32) ? j : j - 32;
    float ang = (float)s * powf(500000.0f, -(float)(2 * i) / 64.0f);
    float out = v * cosf(ang) + ((j < 32) ? -vp : vp) * sinf(ang);
    Qr[((size_t)(b * NH + h) * SEQ + s) * HD + j] = (_Float16)out;
  } else if (c < NH * HD + NKV * HD) {     // K path, rope
    int cc = c - NH * HD;
    int hk = cc / HD, j = cc % HD;
    int jp = (j < 32) ? j + 32 : j - 32;
    float vp = (float)qkv[(size_t)tok * QKVN + NH * HD + hk * HD + jp];
    int i = (j < 32) ? j : j - 32;
    float ang = (float)s * powf(500000.0f, -(float)(2 * i) / 64.0f);
    float out = v * cosf(ang) + ((j < 32) ? -vp : vp) * sinf(ang);
    Kr[((size_t)(b * NKV + hk) * SEQ + s) * HD + j] = (_Float16)out;
  } else {                                 // V path, transpose to [d][S]
    int cc = c - NH * HD - NKV * HD;
    int hk = cc / HD, j = cc % HD;
    Vt[((size_t)(b * NKV + hk) * HD + j) * SEQ + s] = (_Float16)v;
  }
}

// ---------------- GEMM: C[MxN] = A[MxK] * B[KxN], A row-major f16, Bt = B^T row-major f16 ----------------
// one wave per 32x64 C tile; 8 wmma per 32-K step
__global__ __launch_bounds__(32) void gemm_wmma(const _Float16* __restrict__ A,
                                                const _Float16* __restrict__ Bt,
                                                _Float16* __restrict__ Cf16,
                                                float* __restrict__ Cf32,
                                                int M, int N, int K) {
  const int lane = threadIdx.x;
  const int n0 = blockIdx.x * 64;
  const int m0 = blockIdx.y * 32;
  v8f acc[2][4];
#pragma unroll
  for (int i = 0; i < 2; ++i)
#pragma unroll
    for (int j = 0; j < 4; ++j) acc[i][j] = (v8f){};

  for (int k0 = 0; k0 < K; k0 += 32) {
    v16h a0 = load_a16(A, K, m0,      k0, lane);
    v16h a1 = load_a16(A, K, m0 + 16, k0, lane);
    v16h b0 = load_b16(Bt, K, n0,      k0, lane);
    v16h b1 = load_b16(Bt, K, n0 + 16, k0, lane);
    v16h b2 = load_b16(Bt, K, n0 + 32, k0, lane);
    v16h b3 = load_b16(Bt, K, n0 + 48, k0, lane);
    acc[0][0] = WMMA_F16(a0, b0, acc[0][0]);
    acc[0][1] = WMMA_F16(a0, b1, acc[0][1]);
    acc[0][2] = WMMA_F16(a0, b2, acc[0][2]);
    acc[0][3] = WMMA_F16(a0, b3, acc[0][3]);
    acc[1][0] = WMMA_F16(a1, b0, acc[1][0]);
    acc[1][1] = WMMA_F16(a1, b1, acc[1][1]);
    acc[1][2] = WMMA_F16(a1, b2, acc[1][2]);
    acc[1][3] = WMMA_F16(a1, b3, acc[1][3]);
  }

  // C/D layout: lanes 0-15 hold rows m..m+7 (col=lane), lanes 16-31 rows m+8..m+15 (col=lane-16)
  const int colb = lane & 15;
  const int rsel = (lane >> 4) << 3;
#pragma unroll
  for (int i = 0; i < 2; ++i)
#pragma unroll
    for (int j = 0; j < 4; ++j)
#pragma unroll
      for (int r = 0; r < 8; ++r) {
        size_t row = (size_t)(m0 + 16 * i + rsel + r);
        size_t col = (size_t)(n0 + 16 * j + colb);
        if (Cf32) Cf32[row * N + col] = acc[i][j][r];
        else      Cf16[row * N + col] = (_Float16)acc[i][j][r];
      }
}

// ---------------- flash attention: one wave per (b, h, 16-query tile) ----------------
__global__ __launch_bounds__(32) void attn_kernel(const _Float16* __restrict__ Qr,
                                                  const _Float16* __restrict__ Kr,
                                                  const _Float16* __restrict__ Vt,
                                                  const int* __restrict__ cu,
                                                  _Float16* __restrict__ Oout) {
  __shared__ alignas(32) _Float16 Plds[16 * 32];
  const int lane = threadIdx.x;
  const int q0 = blockIdx.x * 16;
  const int h  = blockIdx.y;
  const int b  = blockIdx.z;
  const int hkv = h >> 2;                  // n_rep = 4
  const _Float16* Qp = Qr + (size_t)(b * NH  + h)   * SEQ * HD;
  const _Float16* Kp = Kr + (size_t)(b * NKV + hkv) * SEQ * HD;
  const _Float16* Vp = Vt + (size_t)(b * NKV + hkv) * HD  * SEQ;

  int c0 = cu[b * 4 + 0]; c0 = c0 < 0 ? 0 : (c0 > SEQ ? SEQ : c0);
  int c1 = cu[b * 4 + 1]; c1 = c1 < 0 ? 0 : (c1 > SEQ ? SEQ : c1);
  int c2 = cu[b * 4 + 2]; c2 = c2 < 0 ? 0 : (c2 > SEQ ? SEQ : c2);
  int c3 = cu[b * 4 + 3]; c3 = c3 < 0 ? 0 : (c3 > SEQ ? SEQ : c3);

  const int colb = lane & 15;
  const int rsel = (lane >> 4) << 3;

  int   segq[8];
  float mrow[8], lrow[8];
#pragma unroll
  for (int r = 0; r < 8; ++r) {
    int qr = q0 + rsel + r;
    segq[r] = (c0 <= qr) + (c1 <= qr) + (c2 <= qr) + (c3 <= qr);
    mrow[r] = -1e30f;
    lrow[r] = 0.0f;
  }

  v16h qa0 = load_a16(Qp, HD, q0, 0,  lane);
  v16h qa1 = load_a16(Qp, HD, q0, 32, lane);

  v8f o0 = (v8f){}, o1 = (v8f){}, o2 = (v8f){}, o3 = (v8f){};

  const int kbEnd = (q0 + 15) >> 5;
  for (int kb = 0; kb <= kbEnd; ++kb) {
    const int kbase = kb << 5;
    // scores: S[16q x 32k] = Q[16x64] * K^T, as two 16x16 tiles chained over head-dim
    v8f s0 = (v8f){}, s1 = (v8f){};
    v16h k00 = load_b16(Kp, HD, kbase,      0,  lane);
    v16h k01 = load_b16(Kp, HD, kbase,      32, lane);
    s0 = WMMA_F16(qa0, k00, s0);
    s0 = WMMA_F16(qa1, k01, s0);
    v16h k10 = load_b16(Kp, HD, kbase + 16, 0,  lane);
    v16h k11 = load_b16(Kp, HD, kbase + 16, 32, lane);
    s1 = WMMA_F16(qa0, k10, s1);
    s1 = WMMA_F16(qa1, k11, s1);

    const int key0 = kbase + colb;
    const int key1 = kbase + 16 + colb;
    const int sk0 = (c0 <= key0) + (c1 <= key0) + (c2 <= key0) + (c3 <= key0);
    const int sk1 = (c0 <= key1) + (c1 <= key1) + (c2 <= key1) + (c3 <= key1);

    float sc[8];
#pragma unroll
    for (int r = 0; r < 8; ++r) {
      const int qr = q0 + rsel + r;
      const bool ok0 = (key0 <= qr) && (sk0 == segq[r]);
      const bool ok1 = (key1 <= qr) && (sk1 == segq[r]);
      float v0 = ok0 ? s0[r] * 0.125f : -1e30f;
      float v1 = ok1 ? s1[r] * 0.125f : -1e30f;
      float rm = fmaxf(v0, v1);
#pragma unroll
      for (int m = 1; m < 16; m <<= 1) rm = fmaxf(rm, __shfl_xor(rm, m, 32));
      const float mn = fmaxf(mrow[r], rm);
      const float e0 = ok0 ? __expf(v0 - mn) : 0.0f;
      const float e1 = ok1 ? __expf(v1 - mn) : 0.0f;
      float rs = e0 + e1;
#pragma unroll
      for (int m = 1; m < 16; m <<= 1) rs += __shfl_xor(rs, m, 32);
      const float so = __expf(mrow[r] - mn);
      lrow[r] = lrow[r] * so + rs;
      mrow[r] = mn;
      sc[r] = so;
      Plds[(rsel + r) * 32 + colb]      = (_Float16)e0;
      Plds[(rsel + r) * 32 + 16 + colb] = (_Float16)e1;
    }
#pragma unroll
    for (int r = 0; r < 8; ++r) { o0[r] *= sc[r]; o1[r] *= sc[r]; o2[r] *= sc[r]; o3[r] *= sc[r]; }
    __syncthreads();

    // reload P in A layout from LDS
    v16h pa;
    {
      const _Float16* pp = Plds + colb * 32 + ((lane >> 4) << 4);
      pa = *(const v16h*)pp;
    }
    // O += P[16x32] * V[32x64], V pre-transposed so B-frag loads are contiguous
    v16h bv0 = load_b16(Vp, SEQ, 0,  kbase, lane);
    v16h bv1 = load_b16(Vp, SEQ, 16, kbase, lane);
    v16h bv2 = load_b16(Vp, SEQ, 32, kbase, lane);
    v16h bv3 = load_b16(Vp, SEQ, 48, kbase, lane);
    o0 = WMMA_F16(pa, bv0, o0);
    o1 = WMMA_F16(pa, bv1, o1);
    o2 = WMMA_F16(pa, bv2, o2);
    o3 = WMMA_F16(pa, bv3, o3);
    __syncthreads();
  }

#pragma unroll
  for (int r = 0; r < 8; ++r) {
    const float inv = 1.0f / lrow[r];
    const size_t base = (size_t)(b * SEQ + q0 + rsel + r) * D_MODEL + h * HD + colb;
    Oout[base + 0]  = (_Float16)(o0[r] * inv);
    Oout[base + 16] = (_Float16)(o1[r] * inv);
    Oout[base + 32] = (_Float16)(o2[r] * inv);
    Oout[base + 48] = (_Float16)(o3[r] * inv);
  }
}

// ---------------- host launcher ----------------
extern "C" void kernel_launch(void* const* d_in, const int* in_sizes, int n_in,
                              void* d_out, int out_size, void* d_ws, size_t ws_size,
                              hipStream_t stream) {
  const float* x  = (const float*)d_in[0];
  const int*   cu = (const int*)d_in[1];
  const float* Wq = (const float*)d_in[2];
  const float* Wk = (const float*)d_in[3];
  const float* Wv = (const float*)d_in[4];
  const float* Wo = (const float*)d_in[5];

  char* ws = (char*)d_ws;
  size_t off = 0;
  auto alloc = [&](size_t bytes) -> void* {
    void* p = ws + off;
    off += (bytes + 255) & ~(size_t)255;
    return p;
  };
  _Float16* xh   = (_Float16*)alloc((size_t)NTOK * D_MODEL * 2);        // x in f16
  _Float16* Wt   = (_Float16*)alloc((size_t)QKVN * D_MODEL * 2);        // [Wq|Wk|Wv]^T f16
  _Float16* WoT  = (_Float16*)alloc((size_t)D_MODEL * D_MODEL * 2);     // Wo^T f16
  _Float16* qkv  = (_Float16*)alloc((size_t)NTOK * QKVN * 2);           // fused QKV output
  _Float16* Qr   = (_Float16*)alloc((size_t)BATCH * NH  * SEQ * HD * 2);
  _Float16* Kr   = (_Float16*)alloc((size_t)BATCH * NKV * SEQ * HD * 2);
  _Float16* Vt   = (_Float16*)alloc((size_t)BATCH * NKV * HD * SEQ * 2);
  _Float16* attn = (_Float16*)alloc((size_t)NTOK * D_MODEL * 2);

  // 1. convert x to f16
  f32_to_f16_kernel<<<(NTOK * D_MODEL) / 256, 256, 0, stream>>>(x, xh, (long)NTOK * D_MODEL);

  // 2. transpose+convert weights (concatenated QKV B^T, and Wo^T)
  dim3 tb(32, 8);
  transpose_f32_f16<<<dim3(D_MODEL / 32, D_MODEL / 32), tb, 0, stream>>>(Wq, Wt, D_MODEL, NH * HD);
  transpose_f32_f16<<<dim3((NKV * HD) / 32, D_MODEL / 32), tb, 0, stream>>>(Wk, Wt + (size_t)(NH * HD) * D_MODEL, D_MODEL, NKV * HD);
  transpose_f32_f16<<<dim3((NKV * HD) / 32, D_MODEL / 32), tb, 0, stream>>>(Wv, Wt + (size_t)(NH * HD + NKV * HD) * D_MODEL, D_MODEL, NKV * HD);
  transpose_f32_f16<<<dim3(D_MODEL / 32, D_MODEL / 32), tb, 0, stream>>>(Wo, WoT, NH * HD, D_MODEL);

  // 3. fused QKV projection: [4096 x 2048] * [2048 x 3072]
  gemm_wmma<<<dim3(QKVN / 64, NTOK / 32), 32, 0, stream>>>(xh, Wt, qkv, nullptr, NTOK, QKVN, D_MODEL);

  // 4. RoPE + layout scatter
  rope_scatter<<<((long)NTOK * QKVN) / 256, 256, 0, stream>>>(qkv, Qr, Kr, Vt);

  // 5. flash attention (segmented causal, GQA)
  attn_kernel<<<dim3(SEQ / 16, NH, BATCH), 32, 0, stream>>>(Qr, Kr, Vt, cu, attn);

  // 6. output projection -> fp32 d_out
  gemm_wmma<<<dim3(D_MODEL / 64, NTOK / 32), 32, 0, stream>>>(attn, WoT, nullptr, (float*)d_out, NTOK, D_MODEL, D_MODEL);
}
